// DecoderGenerator_56925496541197
// MI455X (gfx1250) — matrix operand order repaired and checked by reading, not compiled, not run on hardware
//
#include <hip/hip_runtime.h>
#include <hip/hip_bf16.h>

// ---------------------------------------------------------------------------
// Problem constants (reference: V,D,H,L,B,S = 32000,512,512,2,64,64)
// ---------------------------------------------------------------------------
#define VSZ   32000
#define DDIM  512
#define HDIM  512
#define NLAY  2
#define BATCH 64
#define SEQ   64
#define TSTEP 63          // S-1
#define G4H   2048        // 4*H
#define K2H   1024        // 2*H
#define NEGV  (-1000000000.0f)

typedef __bf16 bf16;
typedef __attribute__((ext_vector_type(16))) __bf16 v16bf;
typedef __attribute__((ext_vector_type(8)))  float  v8f;

union ABFrag { v16bf v; uint4 q[2]; };

// ---------------------------------------------------------------------------
// Generic bf16 WMMA GEMM:  out[M x N] = A1[M x K1] * B1[N x K1]^T
//                                      (+ A2[M x K2] * B2[N x K2]^T) (+ bias)
// Block = 256 threads = 8 waves arranged 4(M) x 2(N); each wave owns a
// 16(M) x (NT*16)(N) strip with NT independent accumulators, so one A
// fragment feeds NT WMMAs (NTx arithmetic intensity on A, NT parallel
// accumulation chains to hide v_wmma latency).
// Block tile: 64(M) x (NT*32)(N).  grid.x = N/(NT*32), grid.y = M/64.
// K must be a multiple of 32.  Emits v_wmma_f32_16x16x32_bf16.
// Fragment layouts follow CDNA5 ISA 7.12.2 (wave32):
//   A lane l: row m = m0+(l&15); halves j<8 -> K=kk+(l>>4)*8+j,
//             j>=8 -> K=kk+16+(l>>4)*8+(j-8)   (two contiguous b128 loads)
//   B lane l: col n = n0+(l&15); same K mapping as A (two b128 loads)
//   D lane l: col n = n0+(l&15); vgpr r -> row m = m0+(l>>4)*8+r
// ---------------------------------------------------------------------------
template <int NT>
__global__ void wmma_gemm_bias(const bf16* __restrict__ A1, const bf16* __restrict__ B1, int K1,
                               const bf16* __restrict__ A2, const bf16* __restrict__ B2, int K2,
                               const float* __restrict__ bias,
                               float* __restrict__ out, int ldo) {
  const int lane = threadIdx.x & 31;
  const int wave = threadIdx.x >> 5;
  const int wm = wave & 3;         // 4 waves along M
  const int wn = wave >> 2;        // 2 waves along N
  const int r  = lane & 15;
  const int hi = lane >> 4;
  const int m0 = blockIdx.y * 64 + wm * 16;
  const int n0 = blockIdx.x * (NT * 32) + wn * (NT * 16);

  v8f acc[NT];
#pragma unroll
  for (int j = 0; j < NT; ++j) acc[j] = (v8f){};

  const bf16* As[2] = {A1, A2};
  const bf16* Bs[2] = {B1, B2};
  const int   Ks[2] = {K1, K2};

  for (int p = 0; p < 2; ++p) {
    const bf16* A = As[p];
    if (A == nullptr) break;
    const bf16* Bw = Bs[p];
    const int K = Ks[p];
    const bf16* ar = A + (size_t)(m0 + r) * K;          // A row (M,K) row-major
    const bf16* br[NT];
#pragma unroll
    for (int j = 0; j < NT; ++j)
      br[j] = Bw + (size_t)(n0 + j * 16 + r) * K;       // B rows (N,K) row-major

    for (int kk = 0; kk < K; kk += 32) {
      ABFrag a;
      a.q[0] = *(const uint4*)(ar + kk + hi * 8);
      a.q[1] = *(const uint4*)(ar + kk + 16 + hi * 8);
#pragma unroll
      for (int j = 0; j < NT; ++j) {
        ABFrag b;
        b.q[0] = *(const uint4*)(br[j] + kk + hi * 8);
        b.q[1] = *(const uint4*)(br[j] + kk + 16 + hi * 8);
        __builtin_prefetch((const void*)(br[j] + kk + 128), 0, 1);  // global_prefetch_b8
        acc[j] = __builtin_amdgcn_wmma_f32_16x16x32_bf16(
            /*neg_a=*/false, a.v, /*neg_b=*/false, b.v,
            /*c_mod=*/(short)0, acc[j], /*reuse_a=*/false, /*reuse_b=*/false);
      }
    }
  }

#pragma unroll
  for (int j = 0; j < NT; ++j) {
    const int n = n0 + j * 16 + r;
    const float bv = bias ? bias[n] : 0.0f;
#pragma unroll
    for (int rr = 0; rr < 8; ++rr) {
      const int m = m0 + hi * 8 + rr;
      out[(size_t)m * ldo + n] = acc[j][rr] + bv;
    }
  }
}

// ---------------------------------------------------------------------------
// f32 -> bf16 conversion
// ---------------------------------------------------------------------------
__global__ void cvt_f32_bf16(const float* __restrict__ in, bf16* __restrict__ out, int n) {
  int i = blockIdx.x * blockDim.x + threadIdx.x;
  if (i < n) out[i] = (bf16)in[i];
}

__global__ void zero_u32(unsigned int* __restrict__ p, int n) {
  int i = blockIdx.x * blockDim.x + threadIdx.x;
  if (i < n) p[i] = 0u;
}

// bias_sum[l*G4H+j] = bih[l][j] + bhh[l][j]
__global__ void bias_sum_kernel(const float* __restrict__ bih, const float* __restrict__ bhh,
                                float* __restrict__ bs) {
  int i = blockIdx.x * blockDim.x + threadIdx.x;
  if (i < NLAY * G4H) bs[i] = bih[i] + bhh[i];
}

// E_bf[t][b][d] = bf16(emb[X[b][t]][d]), t in [0,T)
__global__ void embed_kernel(const int* __restrict__ X, const float* __restrict__ emb,
                             bf16* __restrict__ Ebf) {
  int i = blockIdx.x * blockDim.x + threadIdx.x;           // T*B*D
  if (i >= TSTEP * BATCH * DDIM) return;
  int d = i & (DDIM - 1);
  int bt = i / DDIM;
  int b = bt & (BATCH - 1);
  int t = bt / BATCH;
  int tok = X[b * SEQ + t];
  Ebf[((size_t)t * BATCH + b) * DDIM + d] = (bf16)emb[(size_t)tok * DDIM + d];
}

// LSTM pointwise gates: g is [B x 4H] with jnp.split order i,f,g,o
__global__ void lstm_gate(const float* __restrict__ g, float* __restrict__ c,
                          float* __restrict__ h, bf16* __restrict__ h_bf) {
  int i = blockIdx.x * blockDim.x + threadIdx.x;           // B*H
  if (i >= BATCH * HDIM) return;
  int b = i >> 9;            // /HDIM
  int j = i & (HDIM - 1);
  const float* gr = g + (size_t)b * G4H;
  float gi = gr[j];
  float gf = gr[HDIM + j];
  float gc = gr[2 * HDIM + j];
  float go = gr[3 * HDIM + j];
  float si = 1.0f / (1.0f + expf(-gi));
  float sf = 1.0f / (1.0f + expf(-gf));
  float so = 1.0f / (1.0f + expf(-go));
  float cn = sf * c[i] + si * tanhf(gc);
  float hn = so * tanhf(cn);
  c[i] = cn;
  h[i] = hn;
  h_bf[i] = (bf16)hn;
}

// scores[b][e] = sum_h tanh(q[b][h] + enc_proj[b][e][h]) * v_w[h]; mask -> NEG
// one wave per (b,e)
__global__ void attn_scores(const float* __restrict__ q, const float* __restrict__ enc_proj,
                            const float* __restrict__ v_w, const unsigned char* __restrict__ mrow,
                            float* __restrict__ scores) {
  int w = blockIdx.x * (blockDim.x >> 5) + (threadIdx.x >> 5);
  if (w >= BATCH * TSTEP) return;
  int b = w / TSTEP;
  int e = w - b * TSTEP;
  int lane = threadIdx.x & 31;
  const float* qp = q + (size_t)b * HDIM;
  const float* ep = enc_proj + ((size_t)b * TSTEP + e) * HDIM;
  float s = 0.0f;
  for (int hh = lane; hh < HDIM; hh += 32) s += tanhf(qp[hh] + ep[hh]) * v_w[hh];
#pragma unroll
  for (int off = 16; off; off >>= 1) s += __shfl_down(s, off, 32);
  if (lane == 0) scores[w] = mrow[e] ? NEGV : s;
}

// softmax over the BATCH axis (faithful to reference's softmax(s3, axis=1))
// grid = TSTEP blocks, block = BATCH threads
__global__ void softmax_batch(const float* __restrict__ scores, float* __restrict__ att) {
  int e = blockIdx.x;
  int b = threadIdx.x;
  float v = scores[b * TSTEP + e];
  __shared__ float sm[BATCH];
  sm[b] = v;
  __syncthreads();
  for (int off = 32; off; off >>= 1) {
    if (b < off) sm[b] = fmaxf(sm[b], sm[b + off]);
    __syncthreads();
  }
  float mx = sm[0];
  __syncthreads();
  float ex = expf(v - mx);
  sm[b] = ex;
  __syncthreads();
  for (int off = 32; off; off >>= 1) {
    if (b < off) sm[b] += sm[b + off];
    __syncthreads();
  }
  att[b * TSTEP + e] = ex / sm[0];
}

// weighted[b][h] = sum_e att[b][e]*enc[b][e][h]; xcat = concat(top, weighted) in bf16
__global__ void weighted_concat(const float* __restrict__ att, const float* __restrict__ enc,
                                const bf16* __restrict__ h1_bf, bf16* __restrict__ xcat) {
  int i = blockIdx.x * blockDim.x + threadIdx.x;           // B*H
  if (i >= BATCH * HDIM) return;
  int b = i >> 9;
  int hh = i & (HDIM - 1);
  const float* ap = att + (size_t)b * TSTEP;
  const float* ep = enc + ((size_t)b * TSTEP) * HDIM + hh;
  float s = 0.0f;
  for (int e = 0; e < TSTEP; ++e) s += ap[e] * ep[(size_t)e * HDIM];
  xcat[(size_t)b * K2H + hh] = h1_bf[i];
  xcat[(size_t)b * K2H + HDIM + hh] = (bf16)s;
}

// per-row online logsumexp over V, then NLL for target token
__global__ void row_nll(const float* __restrict__ logits, const int* __restrict__ X, int t,
                        float* __restrict__ nll, float* __restrict__ validf) {
  int b = blockIdx.x;
  const float* row = logits + (size_t)b * VSZ;
  float m = -INFINITY, s = 0.0f;
  for (int j = threadIdx.x; j < VSZ; j += 256) {
    float x = row[j];
    if (x > m) { s = s * expf(m - x) + 1.0f; m = x; }
    else       { s += expf(x - m); }
  }
  __shared__ float sm[256], ss[256];
  sm[threadIdx.x] = m;
  ss[threadIdx.x] = s;
  __syncthreads();
  for (int off = 128; off; off >>= 1) {
    if (threadIdx.x < off) {
      float m2 = sm[threadIdx.x + off], s2 = ss[threadIdx.x + off];
      float M = fmaxf(sm[threadIdx.x], m2);
      ss[threadIdx.x] = ss[threadIdx.x] * expf(sm[threadIdx.x] - M) + s2 * expf(m2 - M);
      sm[threadIdx.x] = M;
    }
    __syncthreads();
  }
  if (threadIdx.x == 0) {
    int tgt = X[b * SEQ + t + 1];
    float lse = sm[0] + logf(ss[0]);
    nll[t * BATCH + b] = lse - row[tgt];
    validf[t * BATCH + b] = (tgt != 0) ? 1.0f : 0.0f;
  }
}

// loss = mean_t( sum_b nll*valid / sum_b valid )
__global__ void finalize_loss(const float* __restrict__ nll, const float* __restrict__ validf,
                              float* __restrict__ out) {
  __shared__ float acc[BATCH];
  int t = threadIdx.x;
  float lt = 0.0f;
  if (t < TSTEP) {
    float num = 0.0f, den = 0.0f;
    for (int b = 0; b < BATCH; ++b) {
      num += nll[t * BATCH + b] * validf[t * BATCH + b];
      den += validf[t * BATCH + b];
    }
    lt = num / den;
  }
  acc[t] = lt;
  __syncthreads();
  for (int off = 32; off; off >>= 1) {
    if (t < off) acc[t] += acc[t + off];
    __syncthreads();
  }
  if (t == 0) out[0] = acc[0] / (float)TSTEP;
}

// ---------------------------------------------------------------------------
// Host-side launch
// ---------------------------------------------------------------------------
static inline size_t al256(size_t x) { return (x + 255) & ~(size_t)255; }

extern "C" void kernel_launch(void* const* d_in, const int* in_sizes, int n_in,
                              void* d_out, int out_size, void* d_ws, size_t ws_size,
                              hipStream_t stream) {
  const int*   X       = (const int*)d_in[0];
  const float* enc     = (const float*)d_in[1];
  const unsigned char* mask = (const unsigned char*)d_in[2];
  const float* emb     = (const float*)d_in[3];
  const float* Wih     = (const float*)d_in[4];
  const float* Whh     = (const float*)d_in[5];
  const float* bih     = (const float*)d_in[6];
  const float* bhh     = (const float*)d_in[7];
  const float* attn_Wh = (const float*)d_in[8];
  const float* attn_We = (const float*)d_in[9];
  const float* attn_b  = (const float*)d_in[10];
  const float* v_w     = (const float*)d_in[11];
  const float* fc_W    = (const float*)d_in[12];
  const float* fc_b    = (const float*)d_in[13];
  float* out = (float*)d_out;

  // ---- workspace carve ----
  char* p = (char*)d_ws;
  size_t off = 0;
  auto carve = [&](size_t bytes) { void* r = p + off; off = al256(off + bytes); return r; };

  bf16*  Wih_bf  = (bf16*)carve((size_t)NLAY * G4H * DDIM * 2);
  bf16*  Whh_bf  = (bf16*)carve((size_t)NLAY * G4H * HDIM * 2);
  bf16*  fcW_bf  = (bf16*)carve((size_t)VSZ * K2H * 2);
  bf16*  Wh_bf   = (bf16*)carve((size_t)HDIM * HDIM * 2);
  bf16*  We_bf   = (bf16*)carve((size_t)HDIM * HDIM * 2);
  bf16*  enc_bf  = (bf16*)carve((size_t)BATCH * TSTEP * HDIM * 2);
  bf16*  E_bf    = (bf16*)carve((size_t)TSTEP * BATCH * DDIM * 2);
  float* enc_proj= (float*)carve((size_t)BATCH * TSTEP * HDIM * 4);
  float* bias_s  = (float*)carve((size_t)NLAY * G4H * 4);
  float* hbuf    = (float*)carve((size_t)NLAY * BATCH * HDIM * 4);   // zeroed below
  float* cbuf    = (float*)carve((size_t)NLAY * BATCH * HDIM * 4);   // zeroed below
  bf16*  h_bf    = (bf16*) carve((size_t)NLAY * BATCH * HDIM * 2);   // zeroed below
  float* gbuf    = (float*)carve((size_t)BATCH * G4H * 4);
  float* qbuf    = (float*)carve((size_t)BATCH * HDIM * 4);
  float* scores  = (float*)carve((size_t)BATCH * TSTEP * 4);
  float* att     = (float*)carve((size_t)BATCH * TSTEP * 4);
  bf16*  xcat    = (bf16*) carve((size_t)BATCH * K2H * 2);
  float* logits  = (float*)carve((size_t)BATCH * VSZ * 4);
  float* nll     = (float*)carve((size_t)TSTEP * BATCH * 4);
  float* validf  = (float*)carve((size_t)TSTEP * BATCH * 4);
  (void)ws_size; (void)n_in; (void)in_sizes; (void)out_size;

  const int TB = 256;
  auto blocks = [&](size_t n) { return (unsigned)((n + TB - 1) / TB); };

  // ---- one-time conversions / precompute ----
  cvt_f32_bf16<<<blocks((size_t)NLAY * G4H * DDIM), TB, 0, stream>>>(Wih, Wih_bf, NLAY * G4H * DDIM);
  cvt_f32_bf16<<<blocks((size_t)NLAY * G4H * HDIM), TB, 0, stream>>>(Whh, Whh_bf, NLAY * G4H * HDIM);
  cvt_f32_bf16<<<blocks((size_t)VSZ * K2H), TB, 0, stream>>>(fc_W, fcW_bf, VSZ * K2H);
  cvt_f32_bf16<<<blocks((size_t)HDIM * HDIM), TB, 0, stream>>>(attn_Wh, Wh_bf, HDIM * HDIM);
  cvt_f32_bf16<<<blocks((size_t)HDIM * HDIM), TB, 0, stream>>>(attn_We, We_bf, HDIM * HDIM);
  cvt_f32_bf16<<<blocks((size_t)BATCH * TSTEP * HDIM), TB, 0, stream>>>(enc, enc_bf, BATCH * TSTEP * HDIM);
  bias_sum_kernel<<<blocks(NLAY * G4H), TB, 0, stream>>>(bih, bhh, bias_s);
  embed_kernel<<<blocks((size_t)TSTEP * BATCH * DDIM), TB, 0, stream>>>(X, emb, E_bf);

  // zero h, c (f32) and h_bf (contiguous carve region)
  {
    size_t zw = ((size_t)NLAY * BATCH * HDIM * 4 * 2 + (size_t)NLAY * BATCH * HDIM * 2) / 4;
    zero_u32<<<blocks(zw), TB, 0, stream>>>((unsigned int*)hbuf, (int)zw);
  }

  // enc_proj = enc @ We^T + attn_b  : M=4032, N=512, K=512  (2-tile waves)
  {
    dim3 g(HDIM / 64, (BATCH * TSTEP) / 64);
    wmma_gemm_bias<2><<<g, TB, 0, stream>>>(enc_bf, We_bf, HDIM, nullptr, nullptr, 0,
                                            attn_b, enc_proj, HDIM);
  }

  // ---- time loop ----
  for (int t = 0; t < TSTEP; ++t) {
    const bf16* inp = E_bf + (size_t)t * BATCH * DDIM;
    for (int l = 0; l < NLAY; ++l) {
      const bf16* A1 = (l == 0) ? inp : (h_bf + 0 * (size_t)BATCH * HDIM);
      // g = inp @ Wih[l]^T + h[l] @ Whh[l]^T + (bih+bhh)[l]   (4-tile waves)
      dim3 g(G4H / 128, 1);
      wmma_gemm_bias<4><<<g, TB, 0, stream>>>(
          A1, Wih_bf + (size_t)l * G4H * DDIM, DDIM,
          h_bf + (size_t)l * BATCH * HDIM, Whh_bf + (size_t)l * G4H * HDIM, HDIM,
          bias_s + (size_t)l * G4H, gbuf, G4H);
      lstm_gate<<<blocks((size_t)BATCH * HDIM), TB, 0, stream>>>(
          gbuf, cbuf + (size_t)l * BATCH * HDIM,
          hbuf + (size_t)l * BATCH * HDIM, h_bf + (size_t)l * BATCH * HDIM);
    }
    const bf16* top_bf = h_bf + (size_t)(NLAY - 1) * BATCH * HDIM;

    // q = top @ attn_Wh^T : M=64, N=512, K=512  (2-tile waves)
    {
      dim3 g(HDIM / 64, 1);
      wmma_gemm_bias<2><<<g, TB, 0, stream>>>(top_bf, Wh_bf, HDIM, nullptr, nullptr, 0,
                                              nullptr, qbuf, HDIM);
    }

    int mrowi = ((t - 1) % TSTEP + TSTEP) % TSTEP;
    attn_scores<<<(BATCH * TSTEP + 3) / 4, 128, 0, stream>>>(
        qbuf, enc_proj, v_w, mask + (size_t)mrowi * TSTEP, scores);
    softmax_batch<<<TSTEP, BATCH, 0, stream>>>(scores, att);
    weighted_concat<<<blocks((size_t)BATCH * HDIM), TB, 0, stream>>>(att, enc, top_bf, xcat);

    // logits = xcat @ fc_W^T + fc_b : M=64, N=32000, K=1024  (4-tile waves)
    {
      dim3 g(VSZ / 128, 1);
      wmma_gemm_bias<4><<<g, TB, 0, stream>>>(xcat, fcW_bf, K2H, nullptr, nullptr, 0,
                                              fc_b, logits, VSZ);
    }
    row_nll<<<BATCH, 256, 0, stream>>>(logits, X, t, nll, validf);
  }

  finalize_loss<<<1, BATCH, 0, stream>>>(nll, validf, out);
}
